// DCNBottleneck_15625091023300
// MI455X (gfx1250) — matrix operand order repaired and checked by reading, not compiled, not run on hardware
//
#include <hip/hip_runtime.h>
#include <hip/hip_bf16.h>

// ---------------------------------------------------------------------------
// Types for CDNA5 WMMA (gfx1250, wave32)
// ---------------------------------------------------------------------------
typedef __attribute__((ext_vector_type(16))) __bf16   v16bf;
typedef __attribute__((ext_vector_type(8)))  float    v8f;
typedef __attribute__((ext_vector_type(4)))  unsigned uv4;

union FragAB {
    v16bf  v;
    uv4    q[2];
    __bf16 h[16];
};
union H8 {
    uv4    q;
    __bf16 h[8];
};

// A-matrix 16x32 bf16 fragment per ISA 7.12.2:
//   lanes 0-15 : M=lane,     h[0..7]=K[k0+0..7],  h[8..15]=K[k0+16..23]
//   lanes 16-31: M=lane-16,  h[0..7]=K[k0+8..15], h[8..15]=K[k0+24..31]
__device__ __forceinline__ v16bf load_a_frag(const __bf16* A, int lda, int row,
                                             int k0, int lane) {
    const __bf16* p = A + (size_t)row * lda + k0 + ((lane & 16) ? 8 : 0);
    FragAB f;
    f.q[0] = *(const uv4*)(p);
    f.q[1] = *(const uv4*)(p + 16);
    return f.v;
}

// B-matrix 32x16 bf16 fragment; B stored transposed as Bt[N][K] row-major:
//   lanes 0-15 : N=lane,     h[0..7]=K[k0+0..7],  h[8..15]=K[k0+16..23]
//   lanes 16-31: N=lane-16,  h[0..7]=K[k0+8..15], h[8..15]=K[k0+24..31]
__device__ __forceinline__ v16bf load_b_frag(const __bf16* Bt, int ldb,
                                             int col_base, int k0, int lane) {
    const __bf16* p = Bt + (size_t)(col_base + (lane & 15)) * ldb + k0 +
                      ((lane & 16) ? 8 : 0);
    FragAB f;
    f.q[0] = *(const uv4*)(p);
    f.q[1] = *(const uv4*)(p + 16);
    return f.v;
}

#define WMMA_BF16(A, B, C) \
    __builtin_amdgcn_wmma_f32_16x16x32_bf16(false, (A), false, (B), (short)0, (C), false, false)

// ---------------------------------------------------------------------------
// Problem constants
// ---------------------------------------------------------------------------
#define NB    4
#define CIN   1024
#define P     256
#define HH    64
#define WW    64
#define HW    4096          // 64*64
#define MROWS 16384         // NB*HW
#define K2    2304          // 9*P

// ---------------------------------------------------------------------------
// Param prep: BN scale/shift
// ---------------------------------------------------------------------------
__global__ void prep_scales(const float* g1, const float* b1, const float* m1, const float* v1,
                            const float* g2, const float* b2, const float* m2, const float* v2,
                            const float* g3, const float* b3, const float* m3, const float* v3,
                            float* sc1, float* sh1, float* sc2, float* sh2,
                            float* sc3, float* sh3) {
    int i = threadIdx.x;
    if (i < P) {
        float s = g1[i] * rsqrtf(v1[i] + 1e-5f);
        sc1[i] = s;
        sh1[i] = b1[i] - m1[i] * s;
        s = g2[i] * rsqrtf(v2[i] + 1e-5f);
        sc2[i] = s;
        sh2[i] = b2[i] - m2[i] * s;
    }
    if (i < 4 * P) {
        float s = g3[i] * rsqrtf(v3[i] + 1e-5f);
        sc3[i] = s;
        sh3[i] = b3[i] - m3[i] * s;
    }
}

// w1 [P][CIN] f32 -> bf16 with BN1 scale folded (already GEMM Bt layout [N][K])
__global__ void fold_w1(const float* w1, const float* sc1, __bf16* w1b) {
    int i = blockIdx.x * 256 + threadIdx.x;   // P*CIN = 262144
    int p = i >> 10;
    w1b[i] = (__bf16)(w1[i] * sc1[p]);
}

// w2 [P][P][3][3] -> w2b[o][t*P + c] bf16, BN2 scale folded
__global__ void fold_w2(const float* w2, const float* sc2, __bf16* w2b) {
    int i = blockIdx.x * 256 + threadIdx.x;   // P*K2 = 589824
    if (i >= P * K2) return;
    int o = i / K2;
    int r = i - o * K2;
    int t = r >> 8;        // tap
    int c = r & 255;       // input channel
    w2b[i] = (__bf16)(w2[(size_t)o * K2 + c * 9 + t] * sc2[o]);
}

// w3 [4P][P] -> bf16 with BN3 scale folded
__global__ void fold_w3(const float* w3, const float* sc3, __bf16* w3b) {
    int i = blockIdx.x * 256 + threadIdx.x;   // 4P*P = 262144
    int o = i >> 8;
    w3b[i] = (__bf16)(w3[i] * sc3[o]);
}

// ---------------------------------------------------------------------------
// NCHW f32 -> NHWC bf16 transpose via LDS tiles (32x32)
// ---------------------------------------------------------------------------
__global__ void transpose_x(const float* __restrict__ x, __bf16* __restrict__ xt) {
    __shared__ float tile[32][33];
    int n   = blockIdx.z;
    int hw0 = blockIdx.x * 32;
    int c0  = blockIdx.y * 32;
    int tx  = threadIdx.x;   // 32
    int ty  = threadIdx.y;   // 8
#pragma unroll
    for (int j = 0; j < 4; ++j) {
        int c = c0 + ty + j * 8;
        tile[ty + j * 8][tx] = x[((size_t)n * CIN + c) * HW + hw0 + tx];
    }
    __syncthreads();
#pragma unroll
    for (int j = 0; j < 4; ++j) {
        int hw = hw0 + ty + j * 8;
        xt[((size_t)n * HW + hw) * CIN + c0 + tx] = (__bf16)tile[tx][ty + j * 8];
    }
}

// ---------------------------------------------------------------------------
// conv1: 1x1, M=16384 x K=1024 x N=256, + BN shift + ReLU -> bf16 NHWC.
// 16(M) x 64(N) strip per wave, 4 accumulators. Explicit ping-pong
// double-buffer (even step count -> single loop exit, no rotation copies).
// ---------------------------------------------------------------------------
__global__ void __launch_bounds__(128, 1)
conv1_gemm(const __bf16* __restrict__ xt, const __bf16* __restrict__ w1b,
           const float* __restrict__ sh1, __bf16* __restrict__ out1) {
    int wave   = blockIdx.x * (blockDim.x >> 5) + (threadIdx.x >> 5);
    int lane   = threadIdx.x & 31;
    int ng     = wave & 3;           // 256/64 col groups
    int tile_m = wave >> 2;          // 16384/16 row tiles
    int row    = tile_m * 16 + (lane & 15);
    int nbase  = ng * 64;

    v8f acc0 = {}, acc1 = {}, acc2 = {}, acc3 = {};
    // set 0 = even k-steps, set 1 = odd k-steps (32 steps total)
    v16bf a0  = load_a_frag(xt, CIN, row, 0, lane);
    v16bf p00 = load_b_frag(w1b, CIN, nbase + 0,  0, lane);
    v16bf p01 = load_b_frag(w1b, CIN, nbase + 16, 0, lane);
    v16bf p02 = load_b_frag(w1b, CIN, nbase + 32, 0, lane);
    v16bf p03 = load_b_frag(w1b, CIN, nbase + 48, 0, lane);
    v16bf a1, p10, p11, p12, p13;
    for (int ks = 0; ks < (CIN / 32) - 2; ks += 2) {
        int k1 = (ks + 1) * 32;
        __builtin_prefetch(xt + (size_t)row * CIN + k1 + 64, 0, 3);
        a1  = load_a_frag(xt, CIN, row, k1, lane);
        p10 = load_b_frag(w1b, CIN, nbase + 0,  k1, lane);
        p11 = load_b_frag(w1b, CIN, nbase + 16, k1, lane);
        p12 = load_b_frag(w1b, CIN, nbase + 32, k1, lane);
        p13 = load_b_frag(w1b, CIN, nbase + 48, k1, lane);
        acc0 = WMMA_BF16(a0, p00, acc0);
        acc1 = WMMA_BF16(a0, p01, acc1);
        acc2 = WMMA_BF16(a0, p02, acc2);
        acc3 = WMMA_BF16(a0, p03, acc3);
        int k2 = (ks + 2) * 32;
        a0  = load_a_frag(xt, CIN, row, k2, lane);
        p00 = load_b_frag(w1b, CIN, nbase + 0,  k2, lane);
        p01 = load_b_frag(w1b, CIN, nbase + 16, k2, lane);
        p02 = load_b_frag(w1b, CIN, nbase + 32, k2, lane);
        p03 = load_b_frag(w1b, CIN, nbase + 48, k2, lane);
        acc0 = WMMA_BF16(a1, p10, acc0);
        acc1 = WMMA_BF16(a1, p11, acc1);
        acc2 = WMMA_BF16(a1, p12, acc2);
        acc3 = WMMA_BF16(a1, p13, acc3);
    }
    {   // tail: steps 30 and 31
        int k1 = CIN - 32;
        a1  = load_a_frag(xt, CIN, row, k1, lane);
        p10 = load_b_frag(w1b, CIN, nbase + 0,  k1, lane);
        p11 = load_b_frag(w1b, CIN, nbase + 16, k1, lane);
        p12 = load_b_frag(w1b, CIN, nbase + 32, k1, lane);
        p13 = load_b_frag(w1b, CIN, nbase + 48, k1, lane);
        acc0 = WMMA_BF16(a0, p00, acc0);
        acc1 = WMMA_BF16(a0, p01, acc1);
        acc2 = WMMA_BF16(a0, p02, acc2);
        acc3 = WMMA_BF16(a0, p03, acc3);
        acc0 = WMMA_BF16(a1, p10, acc0);
        acc1 = WMMA_BF16(a1, p11, acc1);
        acc2 = WMMA_BF16(a1, p12, acc2);
        acc3 = WMMA_BF16(a1, p13, acc3);
    }

    int half = (lane >> 4) & 1;
    v8f accs[4] = {acc0, acc1, acc2, acc3};
#pragma unroll
    for (int j = 0; j < 4; ++j) {
        int   col = nbase + j * 16 + (lane & 15);
        float sh  = sh1[col];
#pragma unroll
        for (int r = 0; r < 8; ++r) {
            float v = accs[j][r] + sh;
            v = v > 0.f ? v : 0.f;
            int rr = tile_m * 16 + r + half * 8;
            out1[(size_t)rr * P + col] = (__bf16)v;
        }
    }
}

// ---------------------------------------------------------------------------
// offset conv: 3x3, 256 -> 18 channels, pad 1, bias; tiny -> plain VALU
// offs layout: [m][18] with channel = 2*tap + {0:dy,1:dx}
// ---------------------------------------------------------------------------
__global__ void offset_conv3x3(const __bf16* __restrict__ out1,
                               const float* __restrict__ w_off,
                               const float* __restrict__ b_off,
                               float* __restrict__ offs) {
    int idx = blockIdx.x * blockDim.x + threadIdx.x;   // 16384*18
    if (idx >= MROWS * 18) return;
    int oc = idx % 18;
    int m  = idx / 18;
    int n  = m >> 12;
    int hw = m & 4095;
    int h  = hw >> 6, w = hw & 63;
    float acc = b_off[oc];
    for (int ty = 0; ty < 3; ++ty) {
        int yy = h - 1 + ty;
        if (yy < 0 || yy >= HH) continue;
        for (int tx = 0; tx < 3; ++tx) {
            int xx = w - 1 + tx;
            if (xx < 0 || xx >= WW) continue;
            const __bf16* p  = out1 + ((size_t)(n * HW + yy * WW + xx)) * P;
            const float*  wp = w_off + (size_t)oc * K2 + ty * 3 + tx;  // stride 9 / chan
            float s = 0.f;
            for (int c = 0; c < P; ++c) s += (float)p[c] * wp[(size_t)c * 9];
            acc += s;
        }
    }
    offs[(size_t)m * 18 + oc] = acc;
}

// ---------------------------------------------------------------------------
// deformable 3x3 as implicit GEMM: M=16384 x K=2304 x N=256.
// Per 32-channel chunk: issue the 4 B-fragment loads FIRST, then build the
// A fragment (bilinear corner gathers -> independent VALU hides B latency),
// then 4 WMMAs (16x64 strip per wave, gather work amortized 4x).
// ---------------------------------------------------------------------------
__device__ __forceinline__ void bilin8(__bf16* dst,
                                       const __bf16* p00, const __bf16* p01,
                                       const __bf16* p10, const __bf16* p11,
                                       float w00, float w01, float w10, float w11) {
    H8 a, b, c, d;
    a.q = *(const uv4*)p00;
    b.q = *(const uv4*)p01;
    c.q = *(const uv4*)p10;
    d.q = *(const uv4*)p11;
#pragma unroll
    for (int j = 0; j < 8; ++j) {
        float v = w00 * (float)a.h[j] + w01 * (float)b.h[j] +
                  w10 * (float)c.h[j] + w11 * (float)d.h[j];
        dst[j] = (__bf16)v;
    }
}

__global__ void __launch_bounds__(128, 1)
deform_gemm(const __bf16* __restrict__ out1,
            const float* __restrict__ offs,
            const __bf16* __restrict__ w2b,
            const float* __restrict__ sh2,
            __bf16* __restrict__ out2) {
    int wave   = blockIdx.x * (blockDim.x >> 5) + (threadIdx.x >> 5);
    int lane   = threadIdx.x & 31;
    int ng     = wave & 3;           // 256/64 col groups
    int tile_m = wave >> 2;          // 16384/16 row tiles
    int nbase  = ng * 64;
    int mrow   = tile_m * 16 + (lane & 15);
    int n_img  = mrow >> 12;
    int hw     = mrow & 4095;
    int hh     = hw >> 6, ww = hw & 63;
    int half   = (lane >> 4) & 1;

    const float* offp = offs + (size_t)mrow * 18;
    size_t       base = (size_t)n_img * HW * P;

    v8f acc0 = {}, acc1 = {}, acc2 = {}, acc3 = {};
    for (int t = 0; t < 9; ++t) {
        int   ty = t / 3, tx = t % 3;
        float dy = offp[2 * t + 0];
        float dx = offp[2 * t + 1];
        float py = (float)(hh - 1 + ty) + dy;
        float px = (float)(ww - 1 + tx) + dx;
        float y0f = floorf(py), x0f = floorf(px);
        float wy1 = py - y0f, wx1 = px - x0f;
        float wy0 = 1.f - wy1, wx0 = 1.f - wx1;
        int   y0 = (int)y0f, x0 = (int)x0f;
        int   y1 = y0 + 1, x1 = x0 + 1;
        float vy0 = (y0 >= 0 && y0 < HH) ? 1.f : 0.f;
        float vy1 = (y1 >= 0 && y1 < HH) ? 1.f : 0.f;
        float vx0 = (x0 >= 0 && x0 < WW) ? 1.f : 0.f;
        float vx1 = (x1 >= 0 && x1 < WW) ? 1.f : 0.f;
        // zero-padding folded into corner weights
        float w00 = wy0 * wx0 * vy0 * vx0;
        float w01 = wy0 * wx1 * vy0 * vx1;
        float w10 = wy1 * wx0 * vy1 * vx0;
        float w11 = wy1 * wx1 * vy1 * vx1;
        int y0c = min(max(y0, 0), HH - 1), y1c = min(max(y1, 0), HH - 1);
        int x0c = min(max(x0, 0), WW - 1), x1c = min(max(x1, 0), WW - 1);
        const __bf16* p00 = out1 + base + (size_t)(y0c * WW + x0c) * P;
        const __bf16* p01 = out1 + base + (size_t)(y0c * WW + x1c) * P;
        const __bf16* p10 = out1 + base + (size_t)(y1c * WW + x0c) * P;
        const __bf16* p11 = out1 + base + (size_t)(y1c * WW + x1c) * P;

        for (int c0 = 0; c0 < P; c0 += 32) {
            int k = t * P + c0;
            // B loads first: latency hidden under the bilinear VALU below
            v16bf b0 = load_b_frag(w2b, K2, nbase + 0,  k, lane);
            v16bf b1 = load_b_frag(w2b, K2, nbase + 16, k, lane);
            v16bf b2 = load_b_frag(w2b, K2, nbase + 32, k, lane);
            v16bf b3 = load_b_frag(w2b, K2, nbase + 48, k, lane);
            FragAB a;
            int cg0 = c0 + half * 8;   // h[0..7]
            int cg1 = cg0 + 16;        // h[8..15]
            bilin8(a.h + 0, p00 + cg0, p01 + cg0, p10 + cg0, p11 + cg0,
                   w00, w01, w10, w11);
            bilin8(a.h + 8, p00 + cg1, p01 + cg1, p10 + cg1, p11 + cg1,
                   w00, w01, w10, w11);
            acc0 = WMMA_BF16(a.v, b0, acc0);
            acc1 = WMMA_BF16(a.v, b1, acc1);
            acc2 = WMMA_BF16(a.v, b2, acc2);
            acc3 = WMMA_BF16(a.v, b3, acc3);
        }
    }
    v8f accs[4] = {acc0, acc1, acc2, acc3};
#pragma unroll
    for (int j = 0; j < 4; ++j) {
        int   col = nbase + j * 16 + (lane & 15);
        float sh  = sh2[col];
#pragma unroll
        for (int r = 0; r < 8; ++r) {
            float v = accs[j][r] + sh;
            v = v > 0.f ? v : 0.f;
            int rr = tile_m * 16 + r + half * 8;
            out2[(size_t)rr * P + col] = (__bf16)v;
        }
    }
}

// ---------------------------------------------------------------------------
// conv3: 1x1, M=16384 x K=256 x N=1024, + BN shift + residual(x) + ReLU
// 16x64 strip per wave, explicit ping-pong; output f32 NCHW into d_out.
// ---------------------------------------------------------------------------
__global__ void __launch_bounds__(128, 1)
conv3_gemm(const __bf16* __restrict__ out2, const __bf16* __restrict__ w3b,
           const float* __restrict__ sh3, const float* __restrict__ x,
           float* __restrict__ out) {
    int wave   = blockIdx.x * (blockDim.x >> 5) + (threadIdx.x >> 5);
    int lane   = threadIdx.x & 31;
    int ng     = wave & 15;          // 1024/64 col groups
    int tile_m = wave >> 4;          // 16384/16 row tiles
    int row    = tile_m * 16 + (lane & 15);
    int nbase  = ng * 64;

    v8f acc0 = {}, acc1 = {}, acc2 = {}, acc3 = {};
    v16bf a0  = load_a_frag(out2, P, row, 0, lane);
    v16bf p00 = load_b_frag(w3b, P, nbase + 0,  0, lane);
    v16bf p01 = load_b_frag(w3b, P, nbase + 16, 0, lane);
    v16bf p02 = load_b_frag(w3b, P, nbase + 32, 0, lane);
    v16bf p03 = load_b_frag(w3b, P, nbase + 48, 0, lane);
    v16bf a1, p10, p11, p12, p13;
    for (int ks = 0; ks < (P / 32) - 2; ks += 2) {
        int k1 = (ks + 1) * 32;
        __builtin_prefetch(out2 + (size_t)row * P + k1 + 64, 0, 3);
        a1  = load_a_frag(out2, P, row, k1, lane);
        p10 = load_b_frag(w3b, P, nbase + 0,  k1, lane);
        p11 = load_b_frag(w3b, P, nbase + 16, k1, lane);
        p12 = load_b_frag(w3b, P, nbase + 32, k1, lane);
        p13 = load_b_frag(w3b, P, nbase + 48, k1, lane);
        acc0 = WMMA_BF16(a0, p00, acc0);
        acc1 = WMMA_BF16(a0, p01, acc1);
        acc2 = WMMA_BF16(a0, p02, acc2);
        acc3 = WMMA_BF16(a0, p03, acc3);
        int k2 = (ks + 2) * 32;
        a0  = load_a_frag(out2, P, row, k2, lane);
        p00 = load_b_frag(w3b, P, nbase + 0,  k2, lane);
        p01 = load_b_frag(w3b, P, nbase + 16, k2, lane);
        p02 = load_b_frag(w3b, P, nbase + 32, k2, lane);
        p03 = load_b_frag(w3b, P, nbase + 48, k2, lane);
        acc0 = WMMA_BF16(a1, p10, acc0);
        acc1 = WMMA_BF16(a1, p11, acc1);
        acc2 = WMMA_BF16(a1, p12, acc2);
        acc3 = WMMA_BF16(a1, p13, acc3);
    }
    {   // tail: last two k-steps
        int k1 = P - 32;
        a1  = load_a_frag(out2, P, row, k1, lane);
        p10 = load_b_frag(w3b, P, nbase + 0,  k1, lane);
        p11 = load_b_frag(w3b, P, nbase + 16, k1, lane);
        p12 = load_b_frag(w3b, P, nbase + 32, k1, lane);
        p13 = load_b_frag(w3b, P, nbase + 48, k1, lane);
        acc0 = WMMA_BF16(a0, p00, acc0);
        acc1 = WMMA_BF16(a0, p01, acc1);
        acc2 = WMMA_BF16(a0, p02, acc2);
        acc3 = WMMA_BF16(a0, p03, acc3);
        acc0 = WMMA_BF16(a1, p10, acc0);
        acc1 = WMMA_BF16(a1, p11, acc1);
        acc2 = WMMA_BF16(a1, p12, acc2);
        acc3 = WMMA_BF16(a1, p13, acc3);
    }

    int half = (lane >> 4) & 1;
    v8f accs[4] = {acc0, acc1, acc2, acc3};
#pragma unroll
    for (int j = 0; j < 4; ++j) {
        int   col = nbase + j * 16 + (lane & 15);   // output channel
        float sh  = sh3[col];
#pragma unroll
        for (int r = 0; r < 8; ++r) {
            int    rr  = tile_m * 16 + r + half * 8;
            int    n   = rr >> 12;
            int    hw  = rr & 4095;
            size_t idx = ((size_t)n * (4 * P) + col) * HW + hw;
            float  v   = accs[j][r] + sh + x[idx];
            out[idx] = v > 0.f ? v : 0.f;
        }
    }
}

// ---------------------------------------------------------------------------
// Host launcher
// ---------------------------------------------------------------------------
extern "C" void kernel_launch(void* const* d_in, const int* in_sizes, int n_in,
                              void* d_out, int out_size, void* d_ws, size_t ws_size,
                              hipStream_t stream) {
    (void)in_sizes; (void)n_in; (void)out_size; (void)ws_size;

    const float* x     = (const float*)d_in[0];
    const float* w1    = (const float*)d_in[1];
    const float* g1    = (const float*)d_in[2];
    const float* b1    = (const float*)d_in[3];
    const float* m1    = (const float*)d_in[4];
    const float* v1    = (const float*)d_in[5];
    const float* w_off = (const float*)d_in[6];
    const float* b_off = (const float*)d_in[7];
    const float* w2    = (const float*)d_in[8];
    const float* g2    = (const float*)d_in[9];
    const float* b2    = (const float*)d_in[10];
    const float* m2    = (const float*)d_in[11];
    const float* v2    = (const float*)d_in[12];
    const float* w3    = (const float*)d_in[13];
    const float* g3    = (const float*)d_in[14];
    const float* b3    = (const float*)d_in[15];
    const float* m3    = (const float*)d_in[16];
    const float* v3    = (const float*)d_in[17];

    char*  ws  = (char*)d_ws;
    size_t pos = 0;
    auto take  = [&](size_t bytes) -> char* {
        char* p = ws + pos;
        pos += (bytes + 255) & ~(size_t)255;
        return p;
    };
    __bf16* xt   = (__bf16*)take((size_t)MROWS * CIN * 2);   // 32 MB
    __bf16* out1 = (__bf16*)take((size_t)MROWS * P * 2);     //  8 MB
    __bf16* out2 = (__bf16*)take((size_t)MROWS * P * 2);     //  8 MB
    float*  offs = (float*) take((size_t)MROWS * 18 * 4);    // 1.1 MB
    __bf16* w1b  = (__bf16*)take((size_t)P * CIN * 2);
    __bf16* w2b  = (__bf16*)take((size_t)P * K2 * 2);
    __bf16* w3b  = (__bf16*)take((size_t)4 * P * P * 2);
    float*  sc1  = (float*)take(P * 4);
    float*  sh1  = (float*)take(P * 4);
    float*  sc2  = (float*)take(P * 4);
    float*  sh2  = (float*)take(P * 4);
    float*  sc3  = (float*)take(4 * P * 4);
    float*  sh3  = (float*)take(4 * P * 4);

    prep_scales<<<1, 1024, 0, stream>>>(g1, b1, m1, v1, g2, b2, m2, v2,
                                        g3, b3, m3, v3,
                                        sc1, sh1, sc2, sh2, sc3, sh3);
    fold_w1<<<1024, 256, 0, stream>>>(w1, sc1, w1b);
    fold_w2<<<2304, 256, 0, stream>>>(w2, sc2, w2b);
    fold_w3<<<1024, 256, 0, stream>>>(w3, sc3, w3b);

    transpose_x<<<dim3(HW / 32, CIN / 32, NB), dim3(32, 8), 0, stream>>>(x, xt);

    // conv1: (16384/16 row tiles) * (256/64 col groups) = 4096 waves
    conv1_gemm<<<1024, 128, 0, stream>>>(xt, w1b, sh1, out1);

    offset_conv3x3<<<2304, 128, 0, stream>>>(out1, w_off, b_off, offs);

    // deform: 4096 waves (16x64 strip each)
    deform_gemm<<<1024, 128, 0, stream>>>(out1, offs, w2b, sh2, out2);

    // conv3: (16384/16) * (1024/64) = 16384 waves
    conv3_gemm<<<4096, 128, 0, stream>>>(out2, w3b, sh3, x, (float*)d_out);
}
